// MultiHeadAttention_8856222564432
// MI455X (gfx1250) — matrix operand order repaired and checked by reading, not compiled, not run on hardware
//
#include <hip/hip_runtime.h>
#include <hip/hip_bf16.h>

// MultiHeadAttention forward for MI455X (gfx1250), bf16 WMMA fast path.
//
// Pipeline:
//   0) convert Wq/Wk/Wv/fc_w f32 -> bf16 (once, tiny)
//   1) proj_kernel x3 : QH = q@Wq^T, KH = k@Wk^T, VHT = (v@Wv^T)^T  (bf16)
//   2) attn_kernel    : TDM-stage Q tile -> scores (WMMA) -> scale -> mask ->
//                       softmax (in-register, wave32 shuffle reductions) ->
//                       write attn f32 output -> P@V (WMMA) -> ctx bf16
//   3) fc_ln_kernel   : TDM double-buffered ctx staging -> ctx@fc_w^T + b
//                       (WMMA) -> LayerNorm -> out f32

#define B_SZ   32
#define NH     8
#define SEQ    512
#define DMODEL 512
#define DHEAD  64
#define NEGV   (-1e10f)
#define EPS_LN 1e-5f

typedef __bf16 bf16;
typedef __attribute__((ext_vector_type(16))) __bf16 v16bf;
typedef __attribute__((ext_vector_type(8)))  float  v8f;
typedef __attribute__((ext_vector_type(4)))  unsigned int v4u;
typedef __attribute__((ext_vector_type(8)))  int v8i;
typedef __attribute__((ext_vector_type(4)))  int v4i;

__device__ __forceinline__ v8f wmma_bf16(v16bf a, v16bf b, v8f c) {
  // 8 args: (neg_a, A, neg_b, B, c_mod, C, reuse_a, reuse_b)
  return __builtin_amdgcn_wmma_f32_16x16x32_bf16(false, a, false, b, (short)0, c,
                                                 false, false);
}

// A fragment (16x32 bf16, contiguous k = 16*(lane/16) + i) gathered from an LDS row-major
// tile with row stride `ld` (elements), k-window starting at `kofs`.
__device__ __forceinline__ v16bf load_a_lds(const bf16* base, int ld, int kofs,
                                            int lane) {
  const bf16* row = base + (size_t)(lane & 15) * ld + kofs;
  const int khi = (lane & 16);  // lanes 16-31 hold K offset +16
  v16bf a;
#pragma unroll
  for (int i = 0; i < 16; ++i) a[i] = row[khi + i];
  return a;
}

// B fragment (32x16 bf16): column n = lane&15 is row (ncol0+n) of a row-major
// matrix `mat` (i.e. we multiply by mat^T). Lanes 16-31 hold K offset +16, so
// each lane reads 16 contiguous bf16 -> one aligned 32-byte vector load.
__device__ __forceinline__ v16bf load_b_rowmajor(const bf16* mat, size_t ld,
                                                 int ncol0, int k0, int lane) {
  const bf16* p = mat + (size_t)(ncol0 + (lane & 15)) * ld + k0 + (lane & 16);
  return *(const v16bf*)p;
}

// --------------------------------------------------------------------------
// Tensor Data Mover: 2D tile load global -> LDS (CDNA5 ISA)
// data_size = 8B units. Optional LDS row padding via D# pad fields so the DMA
// engine itself produces the bank-conflict-avoiding strides.
//   pad_iv : encoded pad interval (v -> pad after 2^(v+1) DWORDs)
//   pad_dw : pad amount in DWORDs (0 = no padding)
// Tracked with TENSORcnt (S_WAIT_TENSORCNT); issue from a single wave.
// This toolchain exposes the 6-arg builtin:
//   (uint32x4 g0, int32x8 g1, int32x4 g2, int32x4 g3, int32x8 ext, i32 cpol)
// --------------------------------------------------------------------------
__device__ __forceinline__ void tdm_load_2d(unsigned lds_addr, const void* gptr,
                                            unsigned tile_d0, unsigned tile_d1,
                                            unsigned tensor_d0, unsigned tensor_d1,
                                            unsigned stride_u,
                                            unsigned pad_iv, unsigned pad_dw) {
  unsigned long long ga = (unsigned long long)gptr;
  v4u g0;
  g0.x = 1u;                                        // count=1, user descriptor
  g0.y = lds_addr;                                  // LDS byte address
  g0.z = (unsigned)ga;                              // global_addr[31:0]
  g0.w = (unsigned)((ga >> 32) & 0x01FFFFFFu) |     // global_addr[56:32]
         0x80000000u;                               // type=2 ("image")
  unsigned flags = (3u << 16);                      // data_size = 8 bytes
  if (pad_dw) flags |= (1u << 20) | (pad_iv << 22) | ((pad_dw - 1u) << 25);
  v8i g1;
  g1[0] = (int)flags;                               // wg_mask=0, no atomics
  g1[1] = (int)(tensor_d0 << 16);                   // tensor_dim0[15:0]
  g1[2] = (int)((tensor_d0 >> 16) | (tensor_d1 << 16));
  g1[3] = (int)((tensor_d1 >> 16) | (tile_d0 << 16));
  g1[4] = (int)(tile_d1 & 0xFFFFu);                 // tile_dim1, tile_dim2=0
  g1[5] = (int)stride_u;                            // tensor_dim0_stride[31:0]
  g1[6] = 0;
  g1[7] = 0;
  v4i z4 = {0, 0, 0, 0};                            // groups 2/3 unused (2D)
  v8i z8 = {0, 0, 0, 0, 0, 0, 0, 0};
  __builtin_amdgcn_tensor_load_to_lds(g0, g1, z4, z4, z8, 0);
}

__device__ __forceinline__ unsigned lds_addr_of(const void* p) {
  return (unsigned)(unsigned long long)(uintptr_t)p;  // flat addr[31:0] = LDS offset
}

__global__ void cvt_bf16_kernel(const float* __restrict__ src,
                                bf16* __restrict__ dst, int n) {
  int i = blockIdx.x * blockDim.x + threadIdx.x;
  if (i < n) dst[i] = (bf16)src[i];
}

// ---------------------------------------------------------------------------
// Projection: Y[b,h,l,d] (or transposed [b,h,d,l]) = sum_c X[b,l,c]*W[h*64+d,c]
// Block: 256 thr (8 waves), computes a 64(L) x 64(O) tile; wave -> 2 WMMA tiles.
// f32 -> bf16 conversion happens during LDS staging (TDM cannot convert).
// ---------------------------------------------------------------------------
template <bool TRANS>
__global__ void proj_kernel(const float* __restrict__ X,
                            const bf16* __restrict__ Wb,
                            bf16* __restrict__ Y) {
  __shared__ __align__(16) bf16 Xs[64][36];  // 64 x 32 staged (+pad)
  const int tid = threadIdx.x, lane = tid & 31, wave = tid >> 5;
  const int ob = blockIdx.x * 64, lb = blockIdx.y * 64, b = blockIdx.z;
  const int mt = wave & 3;            // L sub-tile
  const int np = (wave >> 2) * 2;     // O sub-tile pair
  const float* Xbase = X + ((size_t)b * SEQ + lb) * DMODEL;
  v8f acc0 = {}, acc1 = {};
#pragma unroll 1
  for (int ks = 0; ks < 16; ++ks) {
    __syncthreads();
#pragma unroll
    for (int p = 0; p < 8; ++p) {  // 64x32 f32 -> bf16, coalesced
      int idx = tid + p * 256;
      int r = idx >> 5, c = idx & 31;
      Xs[r][c] = (bf16)Xbase[(size_t)r * DMODEL + ks * 32 + c];
    }
    __syncthreads();
    v16bf a  = load_a_lds(&Xs[mt * 16][0], 36, 0, lane);
    v16bf b0 = load_b_rowmajor(Wb, DMODEL, ob + np * 16,       ks * 32, lane);
    v16bf b1 = load_b_rowmajor(Wb, DMODEL, ob + (np + 1) * 16, ks * 32, lane);
    acc0 = wmma_bf16(a, b0, acc0);
    acc1 = wmma_bf16(a, b1, acc1);
  }
  const int n = lane & 15, mhi = (lane & 16) >> 1;
#pragma unroll
  for (int t = 0; t < 2; ++t) {
    v8f acc = t ? acc1 : acc0;
    int o = ob + (np + t) * 16 + n;
    int h = o >> 6, d = o & 63;
#pragma unroll
    for (int r = 0; r < 8; ++r) {
      int l = lb + mt * 16 + mhi + r;
      if (!TRANS)
        Y[(((size_t)b * NH + h) * SEQ + l) * DHEAD + d] = (bf16)acc[r];
      else
        Y[(((size_t)b * NH + h) * DHEAD + d) * SEQ + l] = (bf16)acc[r];
    }
  }
}

// ---------------------------------------------------------------------------
// Attention: per (b,h,32 q-rows). 8 waves; wave = (16-row strip s) x (128-col
// strip ch). Q tile staged by TDM; scores kept in registers (C layout);
// softmax via wave32 shuffles + cross-wave LDS partials; attn f32 written
// head-major; then P@V via WMMA.
// ---------------------------------------------------------------------------
__global__ void attn_kernel(const bf16* __restrict__ QH,
                            const bf16* __restrict__ KH,
                            const bf16* __restrict__ VHT,
                            const int* __restrict__ mask,
                            float* __restrict__ attn_out,
                            bf16* __restrict__ ctx) {
  __shared__ __align__(16) bf16 QHs[32][72];   // 128B rows + 16B TDM pad
  __shared__ __align__(16) bf16 Pb[32][520];
  __shared__ float redA[32][4];
  __shared__ float redB[32][4];
  const int tid = threadIdx.x, lane = tid & 31, wave = tid >> 5;
  const int qb = blockIdx.x * 32;
  const int bh = blockIdx.y, b = bh >> 3, h = bh & 7;
  const int s = wave & 1, ch = wave >> 1;
  const bf16* QHbase = QH + (((size_t)b * NH + h) * SEQ + qb) * DHEAD;
  const bf16* KHbase = KH + ((size_t)b * NH + h) * SEQ * DHEAD;
  const bf16* VTbase = VHT + ((size_t)b * NH + h) * DHEAD * SEQ;

  if (wave == 0) {
    // 32 rows x 128B, padded to 144B LDS rows by the TDM (pad 4 DWORDs
    // after every 32 DWORDs).
    tdm_load_2d(lds_addr_of(&QHs[0][0]), QHbase,
                /*tile_d0=*/16, /*tile_d1=*/32,
                /*tensor_d0=*/16, /*tensor_d1=*/32, /*stride_u=*/16,
                /*pad_iv=*/4, /*pad_dw=*/4);
    __builtin_amdgcn_s_wait_tensorcnt(0);
  }
  __syncthreads();

  // ---- scores: 16 x 128 strip, K = DHEAD = 64 (2 WMMA per tile) ----
  v16bf a0 = load_a_lds(&QHs[s * 16][0], 72, 0, lane);
  v16bf a1 = load_a_lds(&QHs[s * 16][0], 72, 32, lane);
  v8f sc[8];
#pragma unroll
  for (int j = 0; j < 8; ++j) {
    int kcol = ch * 128 + j * 16;
    v16bf b0 = load_b_rowmajor(KHbase, DHEAD, kcol, 0, lane);
    v16bf b1 = load_b_rowmajor(KHbase, DHEAD, kcol, 32, lane);
    v8f c = {};
    c = wmma_bf16(a0, b0, c);
    c = wmma_bf16(a1, b1, c);
    sc[j] = c;
  }

  const float inv_temp = 0.125f;  // 1/sqrt(D_K)
  const int n = lane & 15, mhi = (lane & 16) >> 1;
#pragma unroll
  for (int j = 0; j < 8; ++j) {   // scale + mask
    int col = ch * 128 + j * 16 + n;
#pragma unroll
    for (int r = 0; r < 8; ++r) {
      int q = qb + s * 16 + mhi + r;
      int mv = mask[((size_t)b * SEQ + q) * SEQ + col];
      sc[j][r] = mv ? NEGV : sc[j][r] * inv_temp;
    }
  }

  // ---- softmax: row max ----
  float rmax[8], rsum[8];
#pragma unroll
  for (int r = 0; r < 8; ++r) {
    float v = sc[0][r];
#pragma unroll
    for (int j = 1; j < 8; ++j) v = fmaxf(v, sc[j][r]);
#pragma unroll
    for (int off = 1; off < 16; off <<= 1) v = fmaxf(v, __shfl_xor(v, off, 32));
    if (n == 0) redA[s * 16 + mhi + r][ch] = v;
  }
  __syncthreads();
#pragma unroll
  for (int r = 0; r < 8; ++r) {
    int row = s * 16 + mhi + r;
    rmax[r] = fmaxf(fmaxf(redA[row][0], redA[row][1]),
                    fmaxf(redA[row][2], redA[row][3]));
  }
  // ---- exp + row sum ----
#pragma unroll
  for (int r = 0; r < 8; ++r) {
    float sum = 0.f;
#pragma unroll
    for (int j = 0; j < 8; ++j) {
      float e = __expf(sc[j][r] - rmax[r]);
      sc[j][r] = e;
      sum += e;
    }
#pragma unroll
    for (int off = 1; off < 16; off <<= 1) sum += __shfl_xor(sum, off, 32);
    if (n == 0) redB[s * 16 + mhi + r][ch] = sum;
  }
  __syncthreads();
#pragma unroll
  for (int r = 0; r < 8; ++r) {
    int row = s * 16 + mhi + r;
    rsum[r] = 1.f / (redB[row][0] + redB[row][1] + redB[row][2] + redB[row][3]);
  }
  // ---- normalize: write attn (f32, [h*B+b, q, k]) + Pb (bf16 for WMMA) ----
  float* attn_base = attn_out + (((size_t)h * B_SZ + b) * SEQ + qb) * SEQ;
#pragma unroll
  for (int j = 0; j < 8; ++j) {
    int col = ch * 128 + j * 16 + n;
#pragma unroll
    for (int r = 0; r < 8; ++r) {
      int rl = s * 16 + mhi + r;
      float p = sc[j][r] * rsum[r];
      attn_base[(size_t)rl * SEQ + col] = p;
      Pb[rl][col] = (bf16)p;
    }
  }
  __syncthreads();

  // ---- out = P @ V : 32 x 64, one 16x16 WMMA tile per wave, K = 512 ----
  const int mt = wave & 1, nt = wave >> 1;
  v8f oc = {};
#pragma unroll 1
  for (int ks = 0; ks < 16; ++ks) {
    v16bf a  = load_a_lds(&Pb[mt * 16][0], 520, ks * 32, lane);
    v16bf bb = load_b_rowmajor(VTbase, SEQ, nt * 16, ks * 32, lane);
    oc = wmma_bf16(a, bb, oc);
  }
  const int d = h * DHEAD + nt * 16 + n;
#pragma unroll
  for (int r = 0; r < 8; ++r) {
    int q = qb + mt * 16 + mhi + r;
    ctx[((size_t)b * SEQ + q) * DMODEL + d] = (bf16)oc[r];
  }
}

// ---------------------------------------------------------------------------
// fc + bias + LayerNorm: 32 rows x 512 cols per block; wave = 16-row strip x
// 128-col strip; registers hold the 16x128 result; LN via shuffle + LDS.
// ctx tiles staged by double-buffered TDM (DMA overlaps the WMMA loop).
// ---------------------------------------------------------------------------
__global__ void fc_ln_kernel(const bf16* __restrict__ ctx,
                             const bf16* __restrict__ Wfc,
                             const float* __restrict__ bias,
                             const float* __restrict__ gamma,
                             const float* __restrict__ beta,
                             float* __restrict__ out) {
  __shared__ __align__(16) bf16 Xs[2][32][36];  // 64B rows + 8B TDM pad
  __shared__ float redA[32][4];
  __shared__ float redB[32][4];
  const int tid = threadIdx.x, lane = tid & 31, wave = tid >> 5;
  const int rb = blockIdx.x * 32;  // row block in [0, B*SEQ)
  const int s = wave & 1, ch = wave >> 1;
  const bf16* Xbase = ctx + (size_t)rb * DMODEL;

  if (wave == 0) {
    tdm_load_2d(lds_addr_of(&Xs[0][0][0]), Xbase,
                /*tile_d0=*/8, /*tile_d1=*/32,
                /*tensor_d0=*/128, /*tensor_d1=*/32, /*stride_u=*/128,
                /*pad_iv=*/3, /*pad_dw=*/2);
  }

  v8f zero = {};
  v8f sc[8];
#pragma unroll
  for (int j = 0; j < 8; ++j) sc[j] = zero;
#pragma unroll 1
  for (int ks = 0; ks < 16; ++ks) {
    if (wave == 0) {
      if (ks < 15) {
        tdm_load_2d(lds_addr_of(&Xs[(ks + 1) & 1][0][0]), Xbase + (ks + 1) * 32,
                    8, 32, 128, 32, 128, 3, 2);
        __builtin_amdgcn_s_wait_tensorcnt(1);  // tile ks complete (in-order)
      } else {
        __builtin_amdgcn_s_wait_tensorcnt(0);
      }
    }
    __syncthreads();
    v16bf a = load_a_lds(&Xs[ks & 1][s * 16][0], 36, 0, lane);
#pragma unroll
    for (int j = 0; j < 8; ++j) {
      v16bf bb = load_b_rowmajor(Wfc, DMODEL, ch * 128 + j * 16, ks * 32, lane);
      sc[j] = wmma_bf16(a, bb, sc[j]);
    }
    __syncthreads();
  }
  const int n = lane & 15, mhi = (lane & 16) >> 1;
#pragma unroll
  for (int j = 0; j < 8; ++j) {  // + bias
    float bv = bias[ch * 128 + j * 16 + n];
#pragma unroll
    for (int r = 0; r < 8; ++r) sc[j][r] += bv;
  }
#pragma unroll
  for (int r = 0; r < 8; ++r) {  // LN partials
    float sm = 0.f, sq = 0.f;
#pragma unroll
    for (int j = 0; j < 8; ++j) { float v = sc[j][r]; sm += v; sq += v * v; }
#pragma unroll
    for (int off = 1; off < 16; off <<= 1) {
      sm += __shfl_xor(sm, off, 32);
      sq += __shfl_xor(sq, off, 32);
    }
    if (n == 0) { redA[s*16 + mhi + r][ch] = sm; redB[s*16 + mhi + r][ch] = sq; }
  }
  __syncthreads();
  float mu[8], rstd[8];
#pragma unroll
  for (int r = 0; r < 8; ++r) {
    int row = s * 16 + mhi + r;
    float sm = redA[row][0] + redA[row][1] + redA[row][2] + redA[row][3];
    float sq = redB[row][0] + redB[row][1] + redB[row][2] + redB[row][3];
    float m = sm * (1.f / DMODEL);
    mu[r] = m;
    rstd[r] = rsqrtf(sq * (1.f / DMODEL) - m * m + EPS_LN);
  }
#pragma unroll
  for (int j = 0; j < 8; ++j) {
    int col = ch * 128 + j * 16 + n;
    float g = gamma[col], bt = beta[col];
#pragma unroll
    for (int r = 0; r < 8; ++r) {
      int row = rb + s * 16 + mhi + r;
      out[(size_t)row * DMODEL + col] = (sc[j][r] - mu[r]) * rstd[r] * g + bt;
    }
  }
}

extern "C" void kernel_launch(void* const* d_in, const int* in_sizes, int n_in,
                              void* d_out, int out_size, void* d_ws,
                              size_t ws_size, hipStream_t stream) {
  (void)in_sizes; (void)n_in; (void)out_size; (void)ws_size;
  const float* q    = (const float*)d_in[0];
  const float* k    = (const float*)d_in[1];
  const float* v    = (const float*)d_in[2];
  const int*   mask = (const int*)d_in[3];
  const float* Wq   = (const float*)d_in[4];
  const float* Wk   = (const float*)d_in[5];
  const float* Wv   = (const float*)d_in[6];
  const float* fc_w = (const float*)d_in[7];
  const float* fc_b = (const float*)d_in[8];
  const float* ln_g = (const float*)d_in[9];
  const float* ln_b = (const float*)d_in[10];

  char* ws = (char*)d_ws;
  const size_t WSZ = (size_t)DMODEL * DMODEL * sizeof(bf16);          // 512 KB
  const size_t HSZ = (size_t)B_SZ * NH * SEQ * DHEAD * sizeof(bf16);  // 16 MB
  bf16* WqB = (bf16*)(ws + 0 * WSZ);
  bf16* WkB = (bf16*)(ws + 1 * WSZ);
  bf16* WvB = (bf16*)(ws + 2 * WSZ);
  bf16* FcB = (bf16*)(ws + 3 * WSZ);
  bf16* QH  = (bf16*)(ws + 4 * WSZ);
  bf16* KH  = (bf16*)(ws + 4 * WSZ + 1 * HSZ);
  bf16* VHT = (bf16*)(ws + 4 * WSZ + 2 * HSZ);
  bf16* CTX = (bf16*)(ws + 4 * WSZ + 3 * HSZ);

  float* out0 = (float*)d_out;
  float* attn = out0 + (size_t)B_SZ * SEQ * DMODEL;

  const int nW = DMODEL * DMODEL;
  cvt_bf16_kernel<<<dim3(nW / 256), 256, 0, stream>>>(Wq,   WqB, nW);
  cvt_bf16_kernel<<<dim3(nW / 256), 256, 0, stream>>>(Wk,   WkB, nW);
  cvt_bf16_kernel<<<dim3(nW / 256), 256, 0, stream>>>(Wv,   WvB, nW);
  cvt_bf16_kernel<<<dim3(nW / 256), 256, 0, stream>>>(fc_w, FcB, nW);

  dim3 pgrid(DMODEL / 64, SEQ / 64, B_SZ);
  proj_kernel<false><<<pgrid, 256, 0, stream>>>(q, WqB, QH);
  proj_kernel<false><<<pgrid, 256, 0, stream>>>(k, WkB, KH);
  proj_kernel<true ><<<pgrid, 256, 0, stream>>>(v, WvB, VHT);

  attn_kernel<<<dim3(SEQ / 32, B_SZ * NH), 256, 0, stream>>>(QH, KH, VHT, mask,
                                                             attn, CTX);

  fc_ln_kernel<<<dim3((B_SZ * SEQ) / 32), 256, 0, stream>>>(CTX, FcB, fc_b,
                                                            ln_g, ln_b, out0);
}
